// BevFeatureEncoder_335007449238
// MI455X (gfx1250) — matrix operand order repaired and checked by reading, not compiled, not run on hardware
//
#include <hip/hip_runtime.h>

// ---------------------------------------------------------------------------
// BEV feature encoder for MI455X (gfx1250, wave32, WMMA).
//   K1: zero grid + occupancy
//   K2: fold BN into weights, shuffle into WMMA B-fragment layout (f16)
//   K3: fused MLP 4->64->128->256 (WMMA) + u32-bitpattern atomic segment-max
//   K4: compression GEMM 256->64 (WMMA) + ReLU + transpose to [B,64,GX,GY]
// ---------------------------------------------------------------------------

#define GXC   480
#define GYC   360
#define GXYC  (GXC * GYC)          // 172800
#define NBATCH 2
#define NSEG  (NBATCH * GXYC)      // 345600
#define NPTS  262144
#define BN_EPS 1e-5f

typedef __attribute__((ext_vector_type(16))) _Float16 v16h;
typedef __attribute__((ext_vector_type(8)))  _Float16 v8h;
typedef __attribute__((ext_vector_type(8)))  float    v8f;
typedef __attribute__((ext_vector_type(2)))  float    v2f;

// ---- workspace byte offsets (all 16B-aligned) ----
static constexpr size_t OFF_GRID = 0;                                  // NSEG*256 f32
static constexpr size_t OFF_OCC  = (size_t)NSEG * 256 * 4;             // NSEG i32
static constexpr size_t OFF_W1B  = OFF_OCC + (size_t)NSEG * 4;         // 256 f32  (4x64 folded, B-frag)
static constexpr size_t OFF_B1   = OFF_W1B + 256 * 4;                  // 64 f32
static constexpr size_t OFF_B2   = OFF_B1  + 64 * 4;                   // 128 f32
static constexpr size_t OFF_B3   = OFF_B2  + 128 * 4;                  // 256 f32
static constexpr size_t OFF_W2B  = OFF_B3  + 256 * 4;                  // 64*128 f16 (B-frag)
static constexpr size_t OFF_W3B  = OFF_W2B + (size_t)64 * 128 * 2;     // 128*256 f16 (B-frag)
static constexpr size_t OFF_WCB  = OFF_W3B + (size_t)128 * 256 * 2;    // 256*64 f16 (B-frag)

// ---------------------------------------------------------------------------
__global__ void __launch_bounds__(256) bev_fill_zero(uint4* __restrict__ p, long n)
{
    long i      = (long)blockIdx.x * blockDim.x + threadIdx.x;
    long stride = (long)gridDim.x * blockDim.x;
    uint4 z = {0u, 0u, 0u, 0u};
    for (; i < n; i += stride) p[i] = z;
}

// ---------------------------------------------------------------------------
// Weight prep: fold BN scale into columns, emit B-fragment layouts.
// f16 B 32x16 (KxN): lane L elem j -> K = 32*kt + 8*(L>>4) + (j&7) + 16*(j>>3), N = 16*t + (L&15)
// f32 B  4x16        lane L elem j -> K = 2*(L>>4) + j,          N = 16*t + (L&15)
// ---------------------------------------------------------------------------
__global__ void __launch_bounds__(256) bev_prep(
    const float* __restrict__ w1, const float* __restrict__ g1, const float* __restrict__ b1,
    const float* __restrict__ m1, const float* __restrict__ v1,
    const float* __restrict__ w2, const float* __restrict__ g2, const float* __restrict__ b2,
    const float* __restrict__ m2, const float* __restrict__ v2,
    const float* __restrict__ w3, const float* __restrict__ g3, const float* __restrict__ b3,
    const float* __restrict__ m3, const float* __restrict__ v3,
    const float* __restrict__ wc,
    float* __restrict__ w1B, float* __restrict__ bias1,
    _Float16* __restrict__ w2B, float* __restrict__ bias2,
    _Float16* __restrict__ w3B, float* __restrict__ bias3,
    _Float16* __restrict__ wcB)
{
    const int tid = threadIdx.x;

    for (int i = tid; i < 256; i += 256) {                     // layer1 (f32 frags)
        int t = i >> 6, L = (i >> 1) & 31, j = i & 1;
        int K = 2 * (L >> 4) + j, N = 16 * t + (L & 15);
        float s = g1[N] * rsqrtf(v1[N] + BN_EPS);
        w1B[i] = w1[K * 64 + N] * s;
    }
    for (int i = tid; i < 64; i += 256) {
        float s = g1[i] * rsqrtf(v1[i] + BN_EPS);
        bias1[i] = b1[i] - m1[i] * s;
    }
    for (int i = tid; i < 64 * 128; i += 256) {                // layer2 (f16 frags)
        int j = i & 15, L = (i >> 4) & 31, kt = (i >> 9) & 1, t = i >> 10;
        int K = 32 * kt + 8 * (L >> 4) + (j & 7) + 16 * (j >> 3), N = 16 * t + (L & 15);
        float s = g2[N] * rsqrtf(v2[N] + BN_EPS);
        w2B[i] = (_Float16)(w2[K * 128 + N] * s);
    }
    for (int i = tid; i < 128; i += 256) {
        float s = g2[i] * rsqrtf(v2[i] + BN_EPS);
        bias2[i] = b2[i] - m2[i] * s;
    }
    for (int i = tid; i < 128 * 256; i += 256) {               // layer3 (f16 frags)
        int j = i & 15, L = (i >> 4) & 31, kt = (i >> 9) & 3, t = i >> 11;
        int K = 32 * kt + 8 * (L >> 4) + (j & 7) + 16 * (j >> 3), N = 16 * t + (L & 15);
        float s = g3[N] * rsqrtf(v3[N] + BN_EPS);
        w3B[i] = (_Float16)(w3[K * 256 + N] * s);
    }
    for (int i = tid; i < 256; i += 256) {
        float s = g3[i] * rsqrtf(v3[i] + BN_EPS);
        bias3[i] = b3[i] - m3[i] * s;
    }
    for (int i = tid; i < 256 * 64; i += 256) {                // compression (f16 frags)
        int j = i & 15, L = (i >> 4) & 31, kt = (i >> 9) & 7, t = i >> 12;
        int K = 32 * kt + 8 * (L >> 4) + (j & 7) + 16 * (j >> 3), N = 16 * t + (L & 15);
        wcB[i] = (_Float16)(wc[K * 64 + N]);
    }
}

// ---------------------------------------------------------------------------
// Fused MLP + segment-max scatter. One wave owns 16 points.
// ---------------------------------------------------------------------------
__global__ void __launch_bounds__(256) bev_mlp_scatter(
    const float* __restrict__ voxels, const int* __restrict__ coors,
    const float* __restrict__ w1B, const float* __restrict__ bias1,
    const _Float16* __restrict__ w2B, const float* __restrict__ bias2,
    const _Float16* __restrict__ w3B, const float* __restrict__ bias3,
    float* __restrict__ grid, int* __restrict__ occ)
{
    __shared__ __align__(16) _Float16 stag[8][16 * 128];   // per-wave transpose tile
    __shared__ int segs[8][16];

    const int wave  = threadIdx.x >> 5;
    const int lane  = threadIdx.x & 31;
    const int mrow  = lane & 15;        // M row (point within tile)
    const int khalf = lane >> 4;        // K-half selector
    const int base  = (blockIdx.x * 8 + wave) * 16;

    if (lane < 16) {
        const int p = base + lane;
        const int bb = coors[p * 3 + 0];
        const int xx = coors[p * 3 + 1];
        const int yy = coors[p * 3 + 2];
        const int s  = bb * GXYC + xx * GYC + yy;
        segs[wave][lane] = s;
        occ[s] = 1;                      // benign race: everyone stores 1
    }

    // -------- layer 1: [16x4]f32 @ [4x64]f32 via V_WMMA_F32_16X16X4_F32 ----
#if __has_builtin(__builtin_amdgcn_wmma_f32_16x16x4_f32)
    {
        // A 16x4 f32: lane L -> M=L&15, elems = K {2*(L>>4), 2*(L>>4)+1}
        v2f a1 = *(const v2f*)(voxels + (size_t)(base + mrow) * 4 + 2 * khalf);
        const v2f* w1p = (const v2f*)w1B;
#pragma unroll
        for (int t = 0; t < 4; ++t) {
            v2f b = w1p[t * 32 + lane];
            v8f c = {};
            c = __builtin_amdgcn_wmma_f32_16x16x4_f32(
                    false, a1, false, b, (short)0, c, false, false);
            float bia = bias1[16 * t + mrow];
#pragma unroll
            for (int e = 0; e < 8; ++e) {
                float h = fmaxf(c[e] + bia, 0.0f);
                stag[wave][(e + 8 * khalf) * 64 + 16 * t + mrow] = (_Float16)h;
            }
        }
    }
#else
    {
        // scalar fallback: each lane does one point, half the channels (K=4)
        const float* vr = voxels + (size_t)(base + mrow) * 4;
        float x0 = vr[0], x1 = vr[1], x2 = vr[2], x3 = vr[3];
#pragma unroll
        for (int c = 0; c < 32; ++c) {
            int ch = khalf * 32 + c;
            int i0 = ((ch >> 4) * 32 + (ch & 15)) * 2;     // K=0 position in w1B
            float h = bias1[ch] + x0 * w1B[i0] + x1 * w1B[i0 + 1]
                                + x2 * w1B[i0 + 32] + x3 * w1B[i0 + 33];
            stag[wave][mrow * 64 + ch] = (_Float16)fmaxf(h, 0.0f);
        }
    }
#endif

    // -------- layer 2: [16x64]f16 @ [64x128]f16 ---------------------------
    v16h a2[2];
#pragma unroll
    for (int kt = 0; kt < 2; ++kt) {
        int col = 32 * kt + 8 * khalf;
        v8h lo = *(const v8h*)&stag[wave][mrow * 64 + col];
        v8h hi = *(const v8h*)&stag[wave][mrow * 64 + col + 16];
        a2[kt] = __builtin_shufflevector(lo, hi, 0,1,2,3,4,5,6,7,8,9,10,11,12,13,14,15);
    }
    const v16h* w2p = (const v16h*)w2B;
    v8f acc2[8];
#pragma unroll
    for (int t = 0; t < 8; ++t) {
        v8f c = {};
#pragma unroll
        for (int kt = 0; kt < 2; ++kt)
            c = __builtin_amdgcn_wmma_f32_16x16x32_f16(
                    false, a2[kt], false, w2p[(t * 2 + kt) * 32 + lane],
                    (short)0, c, false, false);
        acc2[t] = c;
    }
#pragma unroll
    for (int t = 0; t < 8; ++t) {
        float bia = bias2[16 * t + mrow];
#pragma unroll
        for (int e = 0; e < 8; ++e) {
            float h = fmaxf(acc2[t][e] + bia, 0.0f);
            stag[wave][(e + 8 * khalf) * 128 + 16 * t + mrow] = (_Float16)h;
        }
    }

    // -------- layer 3: [16x128]f16 @ [128x256]f16 + atomic segment-max ----
    v16h a3[4];
#pragma unroll
    for (int kt = 0; kt < 4; ++kt) {
        int col = 32 * kt + 8 * khalf;
        v8h lo = *(const v8h*)&stag[wave][mrow * 128 + col];
        v8h hi = *(const v8h*)&stag[wave][mrow * 128 + col + 16];
        a3[kt] = __builtin_shufflevector(lo, hi, 0,1,2,3,4,5,6,7,8,9,10,11,12,13,14,15);
    }
    const v16h* w3p = (const v16h*)w3B;
    for (int half = 0; half < 2; ++half) {                 // cap live accumulators
#pragma unroll
        for (int t8 = 0; t8 < 8; ++t8) {
            const int t = half * 8 + t8;
            v8f c = {};
#pragma unroll
            for (int kt = 0; kt < 4; ++kt)
                c = __builtin_amdgcn_wmma_f32_16x16x32_f16(
                        false, a3[kt], false, w3p[(t * 4 + kt) * 32 + lane],
                        (short)0, c, false, false);
            const int ch = 16 * t + mrow;
            const float bia = bias3[ch];
#pragma unroll
            for (int e = 0; e < 8; ++e) {
                float h = fmaxf(c[e] + bia, 0.0f);         // >= 0 always
                int s = segs[wave][e + 8 * khalf];
                // non-negative floats compare monotonically as u32 bit patterns
                atomicMax((unsigned int*)&grid[(size_t)s * 256 + ch],
                          __float_as_uint(h));
            }
        }
    }
}

// ---------------------------------------------------------------------------
// Compression: [16 cells x 256]f16 @ [256x64]f16, ReLU, gate by occupancy,
// write transposed [B,64,GX,GY]. One wave owns 16 cells.
// ---------------------------------------------------------------------------
__global__ void __launch_bounds__(256) bev_compress(
    const float* __restrict__ grid, const int* __restrict__ occ,
    const _Float16* __restrict__ wcB, const float* __restrict__ bc,
    float* __restrict__ out)
{
    __shared__ int occs[8][16];
    const int wave  = threadIdx.x >> 5;
    const int lane  = threadIdx.x & 31;
    const int mrow  = lane & 15;
    const int khalf = lane >> 4;
    const int cellbase = (blockIdx.x * 8 + wave) * 16;

    if (lane < 16) occs[wave][lane] = occ[cellbase + lane];

    // build A fragments directly from the grid row of this lane's cell
    const float* row = grid + (size_t)(cellbase + mrow) * 256;
    v16h a[8];
#pragma unroll
    for (int kt = 0; kt < 8; ++kt) {
        const int col = 32 * kt + 8 * khalf;
        const float* q = row + col;
        v16h t;
#pragma unroll
        for (int j = 0; j < 8; ++j) t[j]     = (_Float16)q[j];
#pragma unroll
        for (int j = 0; j < 8; ++j) t[8 + j] = (_Float16)q[16 + j];
        a[kt] = t;
    }

    const v16h* wcp = (const v16h*)wcB;
#pragma unroll
    for (int t = 0; t < 4; ++t) {
        v8f c = {};
#pragma unroll
        for (int kt = 0; kt < 8; ++kt)
            c = __builtin_amdgcn_wmma_f32_16x16x32_f16(
                    false, a[kt], false, wcp[(t * 8 + kt) * 32 + lane],
                    (short)0, c, false, false);
        const int ch  = 16 * t + mrow;
        const float bia = bc[ch];
#pragma unroll
        for (int e = 0; e < 8; ++e) {
            const int p = e + 8 * khalf;
            const int s = cellbase + p;
            float v = occs[wave][p] ? fmaxf(c[e] + bia, 0.0f) : 0.0f;
            const int bidx = s / GXYC;
            const int rem  = s - bidx * GXYC;              // = x*GY + y
            out[(size_t)bidx * (64 * GXYC) + (size_t)ch * GXYC + rem] = v;
        }
    }
}

// ---------------------------------------------------------------------------
extern "C" void kernel_launch(void* const* d_in, const int* in_sizes, int n_in,
                              void* d_out, int out_size, void* d_ws, size_t ws_size,
                              hipStream_t stream)
{
    const float* voxels = (const float*)d_in[0];
    const int*   coors  = (const int*)d_in[1];
    const float* w1 = (const float*)d_in[3];
    const float* g1 = (const float*)d_in[4];
    const float* b1 = (const float*)d_in[5];
    const float* m1 = (const float*)d_in[6];
    const float* v1 = (const float*)d_in[7];
    const float* w2 = (const float*)d_in[8];
    const float* g2 = (const float*)d_in[9];
    const float* b2 = (const float*)d_in[10];
    const float* m2 = (const float*)d_in[11];
    const float* v2 = (const float*)d_in[12];
    const float* w3 = (const float*)d_in[13];
    const float* g3 = (const float*)d_in[14];
    const float* b3 = (const float*)d_in[15];
    const float* m3 = (const float*)d_in[16];
    const float* v3 = (const float*)d_in[17];
    const float* wc = (const float*)d_in[18];
    const float* bc = (const float*)d_in[19];

    char* ws = (char*)d_ws;
    float*     grid  = (float*)(ws + OFF_GRID);
    int*       occ   = (int*)(ws + OFF_OCC);
    float*     w1B   = (float*)(ws + OFF_W1B);
    float*     bias1 = (float*)(ws + OFF_B1);
    float*     bias2 = (float*)(ws + OFF_B2);
    float*     bias3 = (float*)(ws + OFF_B3);
    _Float16*  w2B   = (_Float16*)(ws + OFF_W2B);
    _Float16*  w3B   = (_Float16*)(ws + OFF_W3B);
    _Float16*  wcB   = (_Float16*)(ws + OFF_WCB);

    // 1) zero grid + occupancy (must reset every launch: atomics accumulate)
    const long nvec16 = (long)(OFF_W1B / 16);   // grid+occ region, 16B chunks
    bev_fill_zero<<<4096, 256, 0, stream>>>((uint4*)ws, nvec16);

    // 2) weight prep (BN fold + fragment shuffle)
    bev_prep<<<1, 256, 0, stream>>>(w1, g1, b1, m1, v1,
                                    w2, g2, b2, m2, v2,
                                    w3, g3, b3, m3, v3, wc,
                                    w1B, bias1, w2B, bias2, w3B, bias3, wcB);

    // 3) fused MLP + segment-max scatter (128 points / block)
    bev_mlp_scatter<<<NPTS / 128, 256, 0, stream>>>(
        voxels, coors, w1B, bias1, w2B, bias2, w3B, bias3, grid, occ);

    // 4) compression GEMM + transpose (128 cells / block)
    bev_compress<<<NSEG / 128, 256, 0, stream>>>(grid, occ, wcB, bc, (float*)d_out);
}
